// _PointnetSAModuleBase_59081570124917
// MI455X (gfx1250) — matrix operand order repaired and checked by reading, not compiled, not run on hardware
//
#include <hip/hip_runtime.h>

// ---------------------------------------------------------------------------
// Problem constants
// ---------------------------------------------------------------------------
#define BB      4
#define NN      16384
#define CIN     64
#define NPT     1024
#define NS      32
#define R2      0.25f      // RADIUS^2
#define K1PAD   96         // 67 channels padded to 3 K-chunks of 32

typedef __attribute__((ext_vector_type(16))) _Float16 v16h;
typedef __attribute__((ext_vector_type(8)))  float    v8f;

union Frag16 { v16h h; unsigned int u[8]; };

// ---------------------------------------------------------------------------
// Kernel 1: furthest point sampling. One block (1024 threads) per batch.
// min-distances live in registers (16 per thread); xyz stays hot in L2.
// ---------------------------------------------------------------------------
__global__ __launch_bounds__(1024) void fps_kernel(const float* __restrict__ xyz,
                                                   int* __restrict__ fps_idx) {
  const int b = blockIdx.x;
  const float* xb = xyz + (long)b * NN * 3;
  __shared__ float s_val[32];
  __shared__ int   s_idx[32];
  __shared__ int   s_far;
  __shared__ float s_cen[3];

  const int tid  = threadIdx.x;
  const int lane = tid & 31;
  const int wv   = tid >> 5;

  float mind[16];
#pragma unroll
  for (int k = 0; k < 16; ++k) mind[k] = 1e10f;

  int far = 0;
  for (int s = 0; s < NPT; ++s) {
    if (tid == 0) {
      fps_idx[b * NPT + s] = far;
      s_cen[0] = xb[far * 3 + 0];
      s_cen[1] = xb[far * 3 + 1];
      s_cen[2] = xb[far * 3 + 2];
    }
    __syncthreads();
    const float cx = s_cen[0], cy = s_cen[1], cz = s_cen[2];

    float best = -1.0f;
    int   bidx = 0;
#pragma unroll
    for (int k = 0; k < 16; ++k) {
      const int p = tid + (k << 10);
      const float dx = xb[p * 3 + 0] - cx;
      const float dy = xb[p * 3 + 1] - cy;
      const float dz = xb[p * 3 + 2] - cz;
      const float d  = dx * dx + dy * dy + dz * dz;
      mind[k] = fminf(mind[k], d);
      if (mind[k] > best) { best = mind[k]; bidx = p; }
    }
    // wave32 argmax reduce (lowest index wins ties, matching jnp.argmax)
#pragma unroll
    for (int off = 16; off > 0; off >>= 1) {
      const float ov = __shfl_down(best, off, 32);
      const int   oi = __shfl_down(bidx, off, 32);
      if (ov > best || (ov == best && oi < bidx)) { best = ov; bidx = oi; }
    }
    if (lane == 0) { s_val[wv] = best; s_idx[wv] = bidx; }
    __syncthreads();
    if (wv == 0) {
      best = s_val[lane];
      bidx = s_idx[lane];
#pragma unroll
      for (int off = 16; off > 0; off >>= 1) {
        const float ov = __shfl_down(best, off, 32);
        const int   oi = __shfl_down(bidx, off, 32);
        if (ov > best || (ov == best && oi < bidx)) { best = ov; bidx = oi; }
      }
      if (lane == 0) s_far = bidx;
    }
    __syncthreads();
    far = s_far;
  }
}

// ---------------------------------------------------------------------------
// Kernel 2: gather new_xyz into the first B*NPT*3 floats of d_out
// ---------------------------------------------------------------------------
__global__ void newxyz_kernel(const float* __restrict__ xyz,
                              const int* __restrict__ fps_idx,
                              float* __restrict__ out) {
  const int i = blockIdx.x * 256 + threadIdx.x;
  if (i >= BB * NPT) return;
  const int b   = i >> 10;
  const int idx = fps_idx[i];
  const float* p = xyz + ((long)b * NN + idx) * 3;
  out[i * 3 + 0] = p[0];
  out[i * 3 + 1] = p[1];
  out[i * 3 + 2] = p[2];
}

// ---------------------------------------------------------------------------
// Kernel 3: ball query. One wave per centroid; keeps the first NS points (by
// index order) inside the radius, padding with first hit (or 0) — matches the
// reference's top_k-over-key-index construction.
// ---------------------------------------------------------------------------
__global__ __launch_bounds__(256) void ballq_kernel(const float* __restrict__ xyz,
                                                    const float* __restrict__ newxyz,
                                                    int* __restrict__ bidx) {
  const int wid  = threadIdx.x >> 5;
  const int lane = threadIdx.x & 31;
  const int g    = blockIdx.x * 8 + wid;   // 0 .. B*NPT-1
  const int b    = g >> 10;
  const float* xb = xyz + (long)b * NN * 3;
  const float cx = newxyz[g * 3 + 0];
  const float cy = newxyz[g * 3 + 1];
  const float cz = newxyz[g * 3 + 2];
  int* out = bidx + g * NS;

  int found = 0;
  for (int base = 0; base < NN && found < NS; base += 32) {
    const int p = base + lane;
    const float dx = xb[p * 3 + 0] - cx;
    const float dy = xb[p * 3 + 1] - cy;
    const float dz = xb[p * 3 + 2] - cz;
    const bool in = (dx * dx + dy * dy + dz * dz) < R2;
    const unsigned long long mask = __ballot(in);
    const int before = __popcll(mask & ((1ull << lane) - 1ull));
    const int pos = found + before;
    if (in && pos < NS) out[pos] = p;
    found += __popcll(mask);
  }
  if (found > NS) found = NS;
  int fill = 0;
  if (found > 0) fill = out[0];
  for (int j = found + lane; j < NS; j += 32) out[j] = fill;
}

// ---------------------------------------------------------------------------
// WMMA fragment loaders (16-bit A 16x32 and B 32x16 VGPR layouts, wave32)
// ---------------------------------------------------------------------------
__device__ __forceinline__ v16h load_a_frag(const _Float16* base, int mbase,
                                            int rowstride, int kbase, int lane) {
  Frag16 f;
  const int mrow  = mbase + (lane & 15);
  const int khalf = (lane < 16) ? 0 : 16;
  const _Float16* p = base + mrow * rowstride + kbase;
#pragma unroll
  for (int g = 0; g < 8; ++g) {
    const int k = khalf + 2 * g;
    f.u[g] = *(const unsigned int*)(p + k);
  }
  return f.h;
}

// B = W^T where W is stored row-major [O][kstride]; B[k][n] = W[n][k]
__device__ __forceinline__ v16h load_b_frag(const _Float16* w, int nbase,
                                            int kstride, int kbase, int lane) {
  Frag16 f;
  const int n     = nbase + (lane & 15);
  const int khalf = (lane < 16) ? 0 : 16;
  const _Float16* p = w + n * kstride + kbase + khalf;
#pragma unroll
  for (int g = 0; g < 8; ++g) f.u[g] = *(const unsigned int*)(p + 2 * g);
  return f.h;
}

// ---------------------------------------------------------------------------
// Kernel 4: gather + 3-layer pointwise MLP (f16 WMMA, f32 accum) + max-pool.
// One wave per centroid, 4 centroids per 128-thread block.
// ---------------------------------------------------------------------------
__global__ __launch_bounds__(128) void mlp_kernel(
    const float* __restrict__ xyz, const float* __restrict__ feats,
    const float* __restrict__ W1, const float* __restrict__ b1,
    const float* __restrict__ W2, const float* __restrict__ b2,
    const float* __restrict__ W3, const float* __restrict__ b3,
    const float* __restrict__ newxyz, const int* __restrict__ ballidx,
    float* __restrict__ outfeat) {
  __shared__ _Float16 sW1[64 * K1PAD];
  __shared__ _Float16 sW2[64 * 64];
  __shared__ _Float16 sW3[128 * 64];
  __shared__ float sb1[64], sb2[64], sb3[128];
  __shared__ _Float16 sh0[4][32 * K1PAD];
  __shared__ _Float16 sh1[4][32 * 64];
  __shared__ _Float16 sh2[4][32 * 64];

  const int tid = threadIdx.x;
  // ---- stage weights as f16 into LDS (zero-pad W1's K to 96) ----
  for (int i = tid; i < 64 * K1PAD; i += 128) {
    const int o = i / K1PAD, k = i % K1PAD;
    sW1[i] = (k < 67) ? (_Float16)W1[o * 67 + k] : (_Float16)0.0f;
  }
  for (int i = tid; i < 64 * 64; i += 128)  sW2[i] = (_Float16)W2[i];
  for (int i = tid; i < 128 * 64; i += 128) sW3[i] = (_Float16)W3[i];
  if (tid < 64) { sb1[tid] = b1[tid]; sb2[tid] = b2[tid]; }
  sb3[tid] = b3[tid];            // tid covers 0..127
  __syncthreads();

  const int wid  = tid >> 5;
  const int lane = tid & 31;
  const int g    = blockIdx.x * 4 + wid;   // centroid id, 0 .. B*NPT-1
  const int b    = g >> 10;
  const int m    = g & (NPT - 1);

  _Float16* h0 = sh0[wid];
  _Float16* h1 = sh1[wid];
  _Float16* h2 = sh2[wid];

  // ---- gather: lane j builds activation row j (3 rel-xyz + 64 features) ----
  {
    const float cx = newxyz[g * 3 + 0];
    const float cy = newxyz[g * 3 + 1];
    const float cz = newxyz[g * 3 + 2];
    const int pidx = ballidx[g * NS + lane];
    const float* xp = xyz + ((long)b * NN + pidx) * 3;
    _Float16* row = h0 + lane * K1PAD;
    row[0] = (_Float16)(xp[0] - cx);
    row[1] = (_Float16)(xp[1] - cy);
    row[2] = (_Float16)(xp[2] - cz);
    const float* fp = feats + (long)b * CIN * NN + pidx;
#pragma unroll
    for (int c = 0; c < CIN; ++c) row[3 + c] = (_Float16)fp[c * NN];
#pragma unroll
    for (int k = 67; k < K1PAD; ++k) row[k] = (_Float16)0.0f;
  }
  // intra-wave LDS: DS ops are in-order within a wave; no block barrier needed

  // ---- layer 1: (32x96) x (96x64) ----
  {
    v16h a[2][3];
#pragma unroll
    for (int mt = 0; mt < 2; ++mt)
#pragma unroll
      for (int kc = 0; kc < 3; ++kc)
        a[mt][kc] = load_a_frag(h0, mt * 16, K1PAD, kc * 32, lane);
#pragma unroll
    for (int mt = 0; mt < 2; ++mt) {
#pragma unroll
      for (int nt = 0; nt < 4; ++nt) {
        const float bv = sb1[nt * 16 + (lane & 15)];
        v8f acc;
#pragma unroll
        for (int q = 0; q < 8; ++q) acc[q] = bv;
#pragma unroll
        for (int kc = 0; kc < 3; ++kc) {
          const v16h bb = load_b_frag(sW1, nt * 16, K1PAD, kc * 32, lane);
          acc = __builtin_amdgcn_wmma_f32_16x16x32_f16(false, a[mt][kc], false, bb,
                                                       (short)0, acc, false, false);
        }
#pragma unroll
        for (int q = 0; q < 8; ++q) {
          const float v  = fmaxf(acc[q], 0.0f);
          const int mm = mt * 16 + q + ((lane < 16) ? 0 : 8);
          const int nn = nt * 16 + (lane & 15);
          h1[mm * 64 + nn] = (_Float16)v;
        }
      }
    }
  }

  // ---- layer 2: (32x64) x (64x64) ----
  {
    v16h a[2][2];
#pragma unroll
    for (int mt = 0; mt < 2; ++mt)
#pragma unroll
      for (int kc = 0; kc < 2; ++kc)
        a[mt][kc] = load_a_frag(h1, mt * 16, 64, kc * 32, lane);
#pragma unroll
    for (int mt = 0; mt < 2; ++mt) {
#pragma unroll
      for (int nt = 0; nt < 4; ++nt) {
        const float bv = sb2[nt * 16 + (lane & 15)];
        v8f acc;
#pragma unroll
        for (int q = 0; q < 8; ++q) acc[q] = bv;
#pragma unroll
        for (int kc = 0; kc < 2; ++kc) {
          const v16h bb = load_b_frag(sW2, nt * 16, 64, kc * 32, lane);
          acc = __builtin_amdgcn_wmma_f32_16x16x32_f16(false, a[mt][kc], false, bb,
                                                       (short)0, acc, false, false);
        }
#pragma unroll
        for (int q = 0; q < 8; ++q) {
          const float v  = fmaxf(acc[q], 0.0f);
          const int mm = mt * 16 + q + ((lane < 16) ? 0 : 8);
          const int nn = nt * 16 + (lane & 15);
          h2[mm * 64 + nn] = (_Float16)v;
        }
      }
    }
  }

  // ---- layer 3: (32x64) x (64x128) + ReLU + max over 32 samples ----
  {
    v16h a[2][2];
#pragma unroll
    for (int mt = 0; mt < 2; ++mt)
#pragma unroll
      for (int kc = 0; kc < 2; ++kc)
        a[mt][kc] = load_a_frag(h2, mt * 16, 64, kc * 32, lane);
#pragma unroll
    for (int nt = 0; nt < 8; ++nt) {
      float nmax = 0.0f;   // ReLU floor
#pragma unroll
      for (int mt = 0; mt < 2; ++mt) {
        const float bv = sb3[nt * 16 + (lane & 15)];
        v8f acc;
#pragma unroll
        for (int q = 0; q < 8; ++q) acc[q] = bv;
#pragma unroll
        for (int kc = 0; kc < 2; ++kc) {
          const v16h bb = load_b_frag(sW3, nt * 16, 64, kc * 32, lane);
          acc = __builtin_amdgcn_wmma_f32_16x16x32_f16(false, a[mt][kc], false, bb,
                                                       (short)0, acc, false, false);
        }
#pragma unroll
        for (int q = 0; q < 8; ++q) nmax = fmaxf(nmax, acc[q]);
      }
      // lanes L and L+16 hold the same N; merge halves
      const float other = __shfl_xor(nmax, 16, 32);
      nmax = fmaxf(nmax, other);
      if (lane < 16) {
        const int c = nt * 16 + lane;
        outfeat[((long)b * 128 + c) * NPT + m] = nmax;
      }
    }
  }
}

// ---------------------------------------------------------------------------
// Host launcher
// ---------------------------------------------------------------------------
extern "C" void kernel_launch(void* const* d_in, const int* in_sizes, int n_in,
                              void* d_out, int out_size, void* d_ws, size_t ws_size,
                              hipStream_t stream) {
  const float* xyz   = (const float*)d_in[0];
  const float* feats = (const float*)d_in[1];
  const float* W1 = (const float*)d_in[2];
  const float* b1 = (const float*)d_in[3];
  const float* W2 = (const float*)d_in[4];
  const float* b2 = (const float*)d_in[5];
  const float* W3 = (const float*)d_in[6];
  const float* b3 = (const float*)d_in[7];

  float* out = (float*)d_out;
  float* out_xyz  = out;                       // B*NPT*3 floats
  float* out_feat = out + BB * NPT * 3;        // B*128*NPT floats

  int* fps  = (int*)d_ws;                      // B*NPT ints
  int* ball = fps + BB * NPT;                  // B*NPT*NS ints

  fps_kernel<<<BB, 1024, 0, stream>>>(xyz, fps);
  newxyz_kernel<<<(BB * NPT + 255) / 256, 256, 0, stream>>>(xyz, fps, out_xyz);
  ballq_kernel<<<(BB * NPT) / 8, 256, 0, stream>>>(xyz, out_xyz, ball);
  mlp_kernel<<<(BB * NPT) / 4, 128, 0, stream>>>(xyz, feats, W1, b1, W2, b2, W3, b3,
                                                 out_xyz, ball, out_feat);
}
